// TemporalGraphLayer_7584912244969
// MI455X (gfx1250) — matrix operand order repaired and checked by reading, not compiled, not run on hardware
//
#include <hip/hip_runtime.h>
#include <hip/hip_bf16.h>
#include <math.h>

typedef __bf16 v16bf __attribute__((ext_vector_type(16)));
typedef __bf16 v8bf  __attribute__((ext_vector_type(8)));
typedef float  v8f   __attribute__((ext_vector_type(8)));

#define DD 128
#define HH 8
#define CC 16
#define CDIM 384
#define NEG_SLOPE 0.2f
#define LN_EPS 1e-5f

// ---- order-preserving float <-> uint mapping for atomicMax-based segment max ----
__device__ __forceinline__ unsigned mapf(float f) {
  unsigned u = __float_as_uint(f);
  return (u & 0x80000000u) ? ~u : (u ^ 0x80000000u);
}
__device__ __forceinline__ float unmapf(unsigned u) {
  return (u & 0x80000000u) ? __uint_as_float(u ^ 0x80000000u) : __uint_as_float(~u);
}

// =====================================================================
// bf16 WMMA GEMM: C[M,Ncol] = epi(A[M,K] @ B[K,Ncol] + bias)
// A: bf16 row-major, stride lda.  Bt: bf16 TRANSPOSED [Ncol,K] row-major.
// One 16x16 tile per wave32; K compile-time -> fully unrolled WMMA chain.
// Per K-step each lane issues exactly 4x 16B loads (2 for A, 2 for Bt).
// EPI: 0 = store f32, 1 = relu -> bf16, 2 = sigmoid -> f32, 3 = tanh -> f32
// =====================================================================
template <int EPI, int K>
__global__ void wmma_gemm_bf16(const __bf16* __restrict__ A, int lda,
                               const __bf16* __restrict__ Bt,
                               const float* __restrict__ bias,
                               float* __restrict__ Cf, __bf16* __restrict__ Cb, int ldc,
                               int M, int Ncol) {
  const int wave = threadIdx.x >> 5;
  const int lane = threadIdx.x & 31;
  const int ntiles = Ncol >> 4;
  const int tile = blockIdx.x * (blockDim.x >> 5) + wave;
  if (tile >= (M >> 4) * ntiles) return;          // wave-uniform: EXEC stays all-ones
  const int row0 = (tile / ntiles) << 4;
  const int col0 = (tile % ntiles) << 4;
  const int half = lane >> 4;
  const int mrow = row0 + (lane & 15);
  const int ncol = col0 + (lane & 15);

  const __bf16* Arow = A + (size_t)mrow * lda;
  const __bf16* Brow = Bt + (size_t)ncol * K;

  v8f acc = {};
#pragma unroll
  for (int k0 = 0; k0 < K; k0 += 32) {
    if (k0 + 32 < K) __builtin_prefetch(Arow + k0 + 32, 0, 3);  // global_prefetch_b8
    v16bf a, b;
    // A 16x32 fragment (ISA layout): lane needs K in [k0+8*half,+8) and [k0+16+8*half,+8)
    ((v8bf*)&a)[0] = *(const v8bf*)(Arow + k0 + half * 8);
    ((v8bf*)&a)[1] = *(const v8bf*)(Arow + k0 + 16 + half * 8);
    // B 32x16 fragment: lane needs K in [k0+8*half,+8) and [k0+16+8*half,+8) of column ncol (= row of Bt)
    ((v8bf*)&b)[0] = *(const v8bf*)(Brow + k0 + half * 8);
    ((v8bf*)&b)[1] = *(const v8bf*)(Brow + k0 + 16 + half * 8);
    acc = __builtin_amdgcn_wmma_f32_16x16x32_bf16(false, a, false, b, (short)0, acc,
                                                  false, false);
  }

  float bv = bias ? bias[ncol] : 0.0f;
#pragma unroll
  for (int r = 0; r < 8; ++r) {
    int row = row0 + r + half * 8;                // C/D layout: VGPR r -> M = r + 8*half
    float v = acc[r] + bv;
    size_t off = (size_t)row * ldc + ncol;
    if (EPI == 0)      Cf[off] = v;
    else if (EPI == 1) Cb[off] = (__bf16)fmaxf(v, 0.0f);
    else if (EPI == 2) Cf[off] = 1.0f / (1.0f + __expf(-v));
    else               Cf[off] = tanhf(v);
  }
}

// =====================================================================
// Prep / elementwise kernels
// =====================================================================
__global__ void k_init_nodes(const float* __restrict__ x, const float* __restrict__ mem,
                             __bf16* __restrict__ comb, float* __restrict__ agg, int total) {
  int idx = blockIdx.x * blockDim.x + threadIdx.x;
  if (idx >= total) return;
  int n = idx >> 7, c = idx & 127;
  comb[(size_t)n * CDIM + c]       = (__bf16)x[idx];
  comb[(size_t)n * CDIM + DD + c]  = (__bf16)mem[idx];
  agg[idx] = 0.0f;
}

__global__ void k_init_mz(unsigned* __restrict__ m, float* __restrict__ z, int total) {
  int idx = blockIdx.x * blockDim.x + threadIdx.x;
  if (idx >= total) return;
  m[idx] = 0u;       // maps to -inf in the order-preserving encoding
  z[idx] = 0.0f;
}

// cast f32 [K,Ncol] -> bf16 transposed [Ncol,K]
__global__ void k_cast_bf16_t(const float* __restrict__ in, __bf16* __restrict__ out,
                              int K, int Ncol) {
  int idx = blockIdx.x * blockDim.x + threadIdx.x;
  if (idx >= K * Ncol) return;
  int k = idx / Ncol, n = idx % Ncol;
  out[(size_t)n * K + k] = (__bf16)in[idx];
}

// per (node, head) attention coefficients a_src / a_dst
__global__ void k_att(const float* __restrict__ xt, const float* __restrict__ att_src,
                      const float* __restrict__ att_dst, float* __restrict__ asrc,
                      float* __restrict__ adst, int total) {
  int idx = blockIdx.x * blockDim.x + threadIdx.x;
  if (idx >= total) return;
  int n = idx >> 3, h = idx & 7;
  const float* xr = xt + (size_t)n * DD + h * CC;
  float a = 0.f, b = 0.f;
#pragma unroll
  for (int c = 0; c < CC; ++c) {
    a += xr[c] * att_src[h * CC + c];
    b += xr[c] * att_dst[h * CC + c];
  }
  asrc[idx] = a;
  adst[idx] = b;
}

// =====================================================================
// Edge softmax passes (self-loops virtualized as edges [E, E+N))
// =====================================================================
__device__ __forceinline__ void edge_ends(const int* es, const int* ed, int e, int E,
                                          int& s, int& d) {
  if (e < E) { s = es[e]; d = ed[e]; }
  else       { s = e - E; d = e - E; }
}

__global__ void k_edge_max(const int* __restrict__ es, const int* __restrict__ ed,
                           const float* __restrict__ asrc, const float* __restrict__ adst,
                           unsigned* __restrict__ mbuf, int E, int Etot) {
  int idx = blockIdx.x * blockDim.x + threadIdx.x;
  if (idx >= Etot * HH) return;
  int e = idx >> 3, h = idx & 7, s, d;
  edge_ends(es, ed, e, E, s, d);
  float v = asrc[s * HH + h] + adst[d * HH + h];
  v = v > 0.f ? v : NEG_SLOPE * v;
  atomicMax(&mbuf[d * HH + h], mapf(v));
}

__global__ void k_edge_sum(const int* __restrict__ es, const int* __restrict__ ed,
                           const float* __restrict__ asrc, const float* __restrict__ adst,
                           const unsigned* __restrict__ mbuf, float* __restrict__ zbuf,
                           int E, int Etot) {
  int idx = blockIdx.x * blockDim.x + threadIdx.x;
  if (idx >= Etot * HH) return;
  int e = idx >> 3, h = idx & 7, s, d;
  edge_ends(es, ed, e, E, s, d);
  float v = asrc[s * HH + h] + adst[d * HH + h];
  v = v > 0.f ? v : NEG_SLOPE * v;
  atomicAdd(&zbuf[d * HH + h], __expf(v - unmapf(mbuf[d * HH + h])));
}

// one wave32 per edge; 4 channels per lane; recompute alpha (cheaper than spilling 27MB)
__global__ void k_edge_scatter(const int* __restrict__ es, const int* __restrict__ ed,
                               const float* __restrict__ asrc, const float* __restrict__ adst,
                               const unsigned* __restrict__ mbuf, const float* __restrict__ zbuf,
                               const float* __restrict__ xt, float* __restrict__ agg,
                               int E, int Etot) {
  int e = blockIdx.x * 4 + (threadIdx.x >> 5);
  if (e >= Etot) return;
  int lane = threadIdx.x & 31, s, d;
  edge_ends(es, ed, e, E, s, d);
  const float* xs = xt + (size_t)s * DD;
  float* ag = agg + (size_t)d * DD;
#pragma unroll
  for (int j = 0; j < 4; ++j) {
    int ch = lane + 32 * j;
    int h = ch >> 4;
    float v = asrc[s * HH + h] + adst[d * HH + h];
    v = v > 0.f ? v : NEG_SLOPE * v;
    float alpha = __expf(v - unmapf(mbuf[d * HH + h])) / zbuf[d * HH + h];
    atomicAdd(&ag[ch], xs[ch] * alpha);
  }
}

// =====================================================================
// residual + LayerNorm; also casts h into combined[:,256:384] as bf16
// =====================================================================
__global__ void k_ln(const float* __restrict__ agg, const float* __restrict__ x,
                     const float* __restrict__ b_gat, const float* __restrict__ ln_w,
                     const float* __restrict__ ln_b, float* __restrict__ hout,
                     __bf16* __restrict__ comb) {
  int n = blockIdx.x, t = threadIdx.x;
  float v = agg[(size_t)n * DD + t] + b_gat[t] + x[(size_t)n * DD + t];
  __shared__ float s1[DD], s2[DD];
  s1[t] = v; s2[t] = v * v;
  __syncthreads();
  for (int s = DD / 2; s > 0; s >>= 1) {
    if (t < s) { s1[t] += s1[t + s]; s2[t] += s2[t + s]; }
    __syncthreads();
  }
  float mu  = s1[0] * (1.0f / DD);
  float var = s2[0] * (1.0f / DD) - mu * mu;
  float hv = (v - mu) * rsqrtf(var + LN_EPS) * ln_w[t] + ln_b[t];
  hout[(size_t)n * DD + t] = hv;
  comb[(size_t)n * CDIM + 2 * DD + t] = (__bf16)hv;
}

// updated_memory = u * memory + (1 - u) * new_memory
__global__ void k_update(const float* __restrict__ u, const float* __restrict__ mem,
                         const float* __restrict__ nm, float* __restrict__ out, int total) {
  int idx = blockIdx.x * blockDim.x + threadIdx.x;
  if (idx >= total) return;
  float uv = u[idx];
  out[idx] = uv * mem[idx] + (1.0f - uv) * nm[idx];
}

// =====================================================================
extern "C" void kernel_launch(void* const* d_in, const int* in_sizes, int n_in,
                              void* d_out, int out_size, void* d_ws, size_t ws_size,
                              hipStream_t stream) {
  const float* x       = (const float*)d_in[0];
  const int*   ei      = (const int*)d_in[1];
  const float* memory  = (const float*)d_in[2];
  const float* W       = (const float*)d_in[3];
  const float* b_gat   = (const float*)d_in[4];
  const float* att_src = (const float*)d_in[5];
  const float* att_dst = (const float*)d_in[6];
  const float* ln_w    = (const float*)d_in[7];
  const float* ln_b    = (const float*)d_in[8];
  const float* Wu1     = (const float*)d_in[9];
  const float* bu1     = (const float*)d_in[10];
  const float* Wu2     = (const float*)d_in[11];
  const float* bu2     = (const float*)d_in[12];
  // d_in[13] = Wr, d_in[14] = br: reference computes reset_gate but never uses it -> skip
  const float* Wn      = (const float*)d_in[15];
  const float* bn      = (const float*)d_in[16];

  const int N = in_sizes[0] / DD;
  const int E = in_sizes[1] / 2;
  const int Etot = E + N;
  const int* es = ei;
  const int* ed = ei + E;

  float* hout = (float*)d_out;
  float* mout = (float*)d_out + (size_t)N * DD;

  // ---- workspace carving (256B aligned) ----
  char* p = (char*)d_ws;
  auto carve = [&](size_t bytes) -> void* {
    void* r = (void*)p;
    p += (bytes + 255) & ~(size_t)255;
    return r;
  };
  float*    xt     = (float*)carve((size_t)N * DD * 4);
  float*    asrc   = (float*)carve((size_t)N * HH * 4);
  float*    adst   = (float*)carve((size_t)N * HH * 4);
  unsigned* mbuf   = (unsigned*)carve((size_t)N * HH * 4);
  float*    zbuf   = (float*)carve((size_t)N * HH * 4);
  float*    agg    = (float*)carve((size_t)N * DD * 4);
  __bf16*   comb   = (__bf16*)carve((size_t)N * CDIM * 2);
  __bf16*   Wbf    = (__bf16*)carve((size_t)DD * DD * 2);     // transposed [DD,DD]
  __bf16*   Wu1bf  = (__bf16*)carve((size_t)CDIM * 256 * 2);  // transposed [256,CDIM]
  __bf16*   Wu2bf  = (__bf16*)carve((size_t)256 * DD * 2);    // transposed [DD,256]
  __bf16*   Wnbf   = (__bf16*)carve((size_t)CDIM * DD * 2);   // transposed [DD,CDIM]
  __bf16*   g1bf   = (__bf16*)carve((size_t)N * 256 * 2);
  float*    ubuf   = (float*)carve((size_t)N * DD * 4);
  float*    nmbuf  = (float*)carve((size_t)N * DD * 4);

  auto cdiv = [](int a, int b) { return (a + b - 1) / b; };
  const int T = 256;

  // prep: bf16 casts (+ weight transpose for contiguous B fragments) + zero accumulators
  k_init_nodes<<<cdiv(N * DD, T), T, 0, stream>>>(x, memory, comb, agg, N * DD);
  k_init_mz<<<cdiv(N * HH, T), T, 0, stream>>>(mbuf, zbuf, N * HH);
  k_cast_bf16_t<<<cdiv(DD * DD, T), T, 0, stream>>>(W, Wbf, DD, DD);
  k_cast_bf16_t<<<cdiv(CDIM * 256, T), T, 0, stream>>>(Wu1, Wu1bf, CDIM, 256);
  k_cast_bf16_t<<<cdiv(256 * DD, T), T, 0, stream>>>(Wu2, Wu2bf, 256, DD);
  k_cast_bf16_t<<<cdiv(CDIM * DD, T), T, 0, stream>>>(Wn, Wnbf, CDIM, DD);

  // GAT linear: xt = x @ W  (A = x section of combined, lda=384)
  {
    int tiles = (N / 16) * (DD / 16);
    wmma_gemm_bf16<0, DD><<<cdiv(tiles, 4), 128, 0, stream>>>(
        comb, CDIM, Wbf, nullptr, xt, nullptr, DD, N, DD);
  }
  k_att<<<cdiv(N * HH, T), T, 0, stream>>>(xt, att_src, att_dst, asrc, adst, N * HH);

  // edge-softmax passes
  k_edge_max<<<cdiv(Etot * HH, T), T, 0, stream>>>(es, ed, asrc, adst, mbuf, E, Etot);
  k_edge_sum<<<cdiv(Etot * HH, T), T, 0, stream>>>(es, ed, asrc, adst, mbuf, zbuf, E, Etot);
  k_edge_scatter<<<cdiv(Etot, 4), 128, 0, stream>>>(es, ed, asrc, adst, mbuf, zbuf, xt, agg,
                                                    E, Etot);

  // residual + LayerNorm -> h (d_out) and combined[:,256:]
  k_ln<<<N, DD, 0, stream>>>(agg, x, b_gat, ln_w, ln_b, hout, comb);

  // MLP: g1 = relu(combined @ Wu1 + bu1) -> bf16
  {
    int tiles = (N / 16) * (256 / 16);
    wmma_gemm_bf16<1, CDIM><<<cdiv(tiles, 4), 128, 0, stream>>>(
        comb, CDIM, Wu1bf, bu1, nullptr, g1bf, 256, N, 256);
  }
  // u = sigmoid(g1 @ Wu2 + bu2)
  {
    int tiles = (N / 16) * (DD / 16);
    wmma_gemm_bf16<2, 256><<<cdiv(tiles, 4), 128, 0, stream>>>(
        g1bf, 256, Wu2bf, bu2, ubuf, nullptr, DD, N, DD);
  }
  // new_memory = tanh(combined @ Wn + bn)
  {
    int tiles = (N / 16) * (DD / 16);
    wmma_gemm_bf16<3, CDIM><<<cdiv(tiles, 4), 128, 0, stream>>>(
        comb, CDIM, Wnbf, bn, nmbuf, nullptr, DD, N, DD);
  }
  // updated_memory
  k_update<<<cdiv(N * DD, T), T, 0, stream>>>(ubuf, memory, nmbuf, mout, N * DD);
}
